// BahdanauAttention_45097156608157
// MI455X (gfx1250) — matrix operand order repaired and checked by reading, not compiled, not run on hardware
//
#include <hip/hip_runtime.h>
#include <hip/hip_bf16.h>

#define B_ 64
#define S_ 4096
#define H_ 512
#define E_ 512
#define NEG_INF_ -1e10f
#define LDA 520        // padded LDS row stride (bf16 elems): 1040B -> 16B-aligned, conflict-free
#define TILES 2        // M-tiles (16 rows each) per wave: 2x B reuse
#define ROWS (TILES*16)

typedef __attribute__((ext_vector_type(16))) __bf16        v16bf;
typedef __attribute__((ext_vector_type(8)))  float         v8f;
typedef __attribute__((ext_vector_type(8)))  unsigned int  v8u;
typedef __attribute__((ext_vector_type(4)))  unsigned int  v4u;

__device__ __forceinline__ unsigned short f2bf(float f) {
  unsigned int u = __float_as_uint(f);
  u += 0x7FFFu + ((u >> 16) & 1u);   // round-to-nearest-even
  return (unsigned short)(u >> 16);
}

// ---------------------------------------------------------------------------
// K0: Wk (E x H, fp32 row-major) -> bf16 pre-swizzled B-operand tiles.
// Layout: WkB[((kk*32 + n)*32 + lane)*16 + j]
//   lane<16 : element j is K = kk*32 + j       , N = n*16 + (lane&15)
//   lane>=16: element j is K = kk*32 + 16 + j  , N = n*16 + (lane&15)
// ---------------------------------------------------------------------------
__global__ __launch_bounds__(256) void prep_wk(const float* __restrict__ Wk,
                                               unsigned short* __restrict__ WkB) {
  int idx  = blockIdx.x * blockDim.x + threadIdx.x;  // 0..262143
  int j    = idx & 15;
  int lane = (idx >> 4) & 31;
  int n    = (idx >> 9) & 31;
  int kk   = idx >> 14;
  int krow = kk * 32 + (lane >> 4) * 16 + j;
  int col  = n * 16 + (lane & 15);
  WkB[idx] = f2bf(Wk[krow * H_ + col]);
}

// ---------------------------------------------------------------------------
// K1: q[b][h] = hidden[b]@Wq[:,h] + bq[h] + bk[h]   (bk folded in)
// ---------------------------------------------------------------------------
__global__ __launch_bounds__(512) void qproj(const float* __restrict__ hidden,
                                             const float* __restrict__ Wq,
                                             const float* __restrict__ bq,
                                             const float* __restrict__ bk,
                                             float* __restrict__ qbuf) {
  __shared__ float hrow[H_];
  int b = blockIdx.x, h = threadIdx.x;
  hrow[h] = hidden[b * H_ + h];
  __syncthreads();
  float a = 0.f;
#pragma unroll 8
  for (int i = 0; i < H_; ++i) a = fmaf(hrow[i], Wq[i * H_ + h], a);
  qbuf[b * H_ + h] = a + bq[h] + bk[h];
}

// ---------------------------------------------------------------------------
// K2: fused  scores[b,s] = tanh(q[b]+enc[b,s]@Wk) . Wv + bv  (+ mask)
// 1 block = 128 thr (4 waves) per 32-row M super-tile (2 WMMA M-tiles/wave).
// All 16 B operands for an N-tile are loaded as one burst into distinct
// registers, then consumed by 32 streaming WMMAs (incremental loadcnt waits).
// ---------------------------------------------------------------------------
__global__ __launch_bounds__(128) void scores_kernel(
    const float* __restrict__ enc, const unsigned short* __restrict__ WkB,
    const float* __restrict__ qbuf, const float* __restrict__ Wv,
    const float* __restrict__ bv, const int* __restrict__ mask,
    float* __restrict__ scores) {
  extern __shared__ __align__(16) char smem[];
  unsigned short* As   = (unsigned short*)smem;           // ROWS x LDA bf16
  float*          sred = (float*)(smem + ROWS * LDA * 2); // ROWS floats

  int tid     = threadIdx.x;
  int rowBase = blockIdx.x * ROWS;   // row in flattened (B*S); 32 | 4096 -> one batch
  int b       = rowBase / S_;
  int sBase   = rowBase % S_;

  if (tid < ROWS) sred[tid] = 0.f;

  // Stage ROWS x 512 fp32 tile -> bf16 LDS (coalesced float4 loads)
  const float* src = enc + (size_t)rowBase * E_;
#pragma unroll 4
  for (int it = 0; it < ROWS * 2; ++it) {    // one 128-float4 row per iteration
    int L   = it * 128 + tid;
    int row = L >> 7;
    int c4  = L & 127;
    float4 v = ((const float4*)(src + (size_t)row * E_))[c4];
    unsigned int lo = (unsigned)f2bf(v.x) | ((unsigned)f2bf(v.y) << 16);
    unsigned int hi = (unsigned)f2bf(v.z) | ((unsigned)f2bf(v.w) << 16);
    unsigned int* dst = (unsigned int*)&As[row * LDA + c4 * 4];
    dst[0] = lo; dst[1] = hi;
  }
  __syncthreads();

  int wave   = tid >> 5;
  int lane   = tid & 31;
  int laneHi = lane >> 4;           // which K-half this lane supplies
  int ncol   = lane & 15;           // C/D column within N-tile
  int m      = lane & 15;           // A row (within tile) for this lane

  float acc[TILES][8];
#pragma unroll
  for (int t = 0; t < TILES; ++t)
#pragma unroll
    for (int i = 0; i < 8; ++i) acc[t][i] = 0.f;

  for (int nn = 0; nn < 8; ++nn) {
    int n = wave * 8 + nn;          // global N-tile 0..31
    const unsigned short* bBase = WkB + ((size_t)(n * 32 + lane)) * 16;
    // elems between kk steps: 32*32*16 = 16384

    // Burst-load ALL 16 B operands into independent registers (32 loads in
    // flight under one clause), so WMMAs below drain them incrementally.
    v16bf ball[16];
#pragma unroll
    for (int kk = 0; kk < 16; ++kk)
      ball[kk] = __builtin_bit_cast(v16bf, *(const v8u*)(bBase + (size_t)kk * 16384));

    v8f c[TILES];
#pragma unroll
    for (int t = 0; t < TILES; ++t) c[t] = (v8f){0.f,0.f,0.f,0.f,0.f,0.f,0.f,0.f};

#pragma unroll
    for (int kk = 0; kk < 16; ++kk) {
#pragma unroll
      for (int t = 0; t < TILES; ++t) {
        const unsigned short* aRow = &As[(t * 16 + m) * LDA];
        const v4u* ap = (const v4u*)(aRow + kk * 32 + laneHi * 16);
        v4u a0 = ap[0];             // K {0..7}/{16..23}
        v4u a1 = ap[1];             // +16B: K {8..15}/{24..31}
        v8u au;
        au[0] = a0[0]; au[1] = a0[1]; au[2] = a0[2]; au[3] = a0[3];
        au[4] = a1[0]; au[5] = a1[1]; au[6] = a1[2]; au[7] = a1[3];
        v16bf amat = __builtin_bit_cast(v16bf, au);
        c[t] = __builtin_amdgcn_wmma_f32_16x16x32_bf16(false, amat, false, ball[kk],
                                                       (short)0, c[t], false, false);
      }
    }
    // epilogue: + q, tanh, * Wv, accumulate over this wave's columns
    float qv = qbuf[b * H_ + n * 16 + ncol];
    float wv = Wv[n * 16 + ncol];
#pragma unroll
    for (int t = 0; t < TILES; ++t)
#pragma unroll
      for (int i = 0; i < 8; ++i) acc[t][i] += tanhf(c[t][i] + qv) * wv;
  }

  // sum across the 16 columns of each lane half (C layout: rows i / i+8)
#pragma unroll
  for (int t = 0; t < TILES; ++t)
#pragma unroll
    for (int i = 0; i < 8; ++i) {
      float v = acc[t][i];
      v += __shfl_xor(v, 1, 32);
      v += __shfl_xor(v, 2, 32);
      v += __shfl_xor(v, 4, 32);
      v += __shfl_xor(v, 8, 32);
      acc[t][i] = v;
    }
  if (lane == 0 || lane == 16) {
    int rb = laneHi * 8;
#pragma unroll
    for (int t = 0; t < TILES; ++t)
#pragma unroll
      for (int i = 0; i < 8; ++i) atomicAdd(&sred[t * 16 + rb + i], acc[t][i]);
  }
  __syncthreads();

  if (tid < ROWS) {
    int s = sBase + tid;
    float sc = sred[tid] + bv[0];
    if (mask[b * S_ + s] == 0) sc = NEG_INF_;
    scores[b * S_ + s] = sc;
  }
}

// ---------------------------------------------------------------------------
// K3: softmax over S per batch row; writes attn directly into d_out
// ---------------------------------------------------------------------------
__global__ __launch_bounds__(256) void softmax_kernel(const float* __restrict__ scores,
                                                      float* __restrict__ attn) {
  __shared__ float red[256];
  int b = blockIdx.x, tid = threadIdx.x;
  const float* row = scores + b * S_;
  float* out = attn + b * S_;

  float m = -3.4e38f;
  for (int s = tid; s < S_; s += 256) m = fmaxf(m, row[s]);
  red[tid] = m; __syncthreads();
  for (int o = 128; o > 0; o >>= 1) { if (tid < o) red[tid] = fmaxf(red[tid], red[tid + o]); __syncthreads(); }
  float mv = red[0]; __syncthreads();

  float sum = 0.f;
  for (int s = tid; s < S_; s += 256) { float e = __expf(row[s] - mv); out[s] = e; sum += e; }
  red[tid] = sum; __syncthreads();
  for (int o = 128; o > 0; o >>= 1) { if (tid < o) red[tid] += red[tid + o]; __syncthreads(); }
  float inv = 1.f / red[0];
  for (int s = tid; s < S_; s += 256) out[s] *= inv;
}

// ---------------------------------------------------------------------------
// K4: context[b,e] = sum_s attn[b,s] * enc[b,s,e]
// ---------------------------------------------------------------------------
__global__ __launch_bounds__(512) void zero_ctx(float* __restrict__ ctx) {
  ctx[blockIdx.x * 512 + threadIdx.x] = 0.f;
}

__global__ __launch_bounds__(512) void context_kernel(const float* __restrict__ enc,
                                                      const float* __restrict__ attn,
                                                      float* __restrict__ ctx) {
  __shared__ float sa[512];
  int b = blockIdx.x;
  int s0 = blockIdx.y * 512;
  int e = threadIdx.x;
  sa[e] = attn[b * S_ + s0 + e];
  __syncthreads();
  const float* ep = enc + ((size_t)b * S_ + s0) * E_;
  float a = 0.f;
#pragma unroll 4
  for (int s = 0; s < 512; ++s) a = fmaf(sa[s], ep[(size_t)s * E_ + e], a);
  atomicAdd(&ctx[b * E_ + e], a);
}

// ---------------------------------------------------------------------------
extern "C" void kernel_launch(void* const* d_in, const int* in_sizes, int n_in,
                              void* d_out, int out_size, void* d_ws, size_t ws_size,
                              hipStream_t stream) {
  const float* hidden = (const float*)d_in[0];
  const float* enc    = (const float*)d_in[1];
  const int*   mask   = (const int*)d_in[2];
  const float* Wq     = (const float*)d_in[3];
  const float* bq     = (const float*)d_in[4];
  const float* Wk     = (const float*)d_in[5];
  const float* bk     = (const float*)d_in[6];
  const float* Wv     = (const float*)d_in[7];
  const float* bv     = (const float*)d_in[8];

  float* out     = (float*)d_out;
  float* context = out;                 // (B, E)
  float* attn    = out + B_ * E_;       // (B, S)

  char* ws = (char*)d_ws;
  unsigned short* WkB   = (unsigned short*)ws;            // 512 KB
  float*          qbuf  = (float*)(ws + 524288);          // 128 KB
  float*          score = (float*)(ws + 655360);          // 1 MB

  size_t smem = (size_t)ROWS * LDA * 2 + ROWS * 4;        // ~33.4 KB dynamic LDS

  prep_wk      <<<1024, 256, 0, stream>>>(Wk, WkB);
  qproj        <<<B_,   512, 0, stream>>>(hidden, Wq, bq, bk, qbuf);
  scores_kernel<<<(B_ * S_) / ROWS, 128, smem, stream>>>(enc, WkB, qbuf, Wv, bv, mask, score);
  softmax_kernel<<<B_, 256, 0, stream>>>(score, attn);
  zero_ctx     <<<B_, 512, 0, stream>>>(context);
  context_kernel<<<dim3(B_, 8), 512, 0, stream>>>(enc, attn, context);
}
